// PolarityLongformer_76673756168220
// MI455X (gfx1250) — compile-verified, matrix operand order checked by reading, not run on hardware
//
#include <hip/hip_runtime.h>
#include <hip/hip_bf16.h>
#include <math.h>

// ---- model dims (compile-time constants from the reference) ----
#define LNUM   12
#define DDIM   768
#define NHEAD  12
#define HDIM   64
#define WWIN   256
#define FFDIM  3072
#define SEQ    4096
#define BATCH  2

typedef unsigned short u16;
typedef __attribute__((ext_vector_type(16))) __bf16           v16bf;
typedef __attribute__((ext_vector_type(16))) unsigned short   v16u;
typedef __attribute__((ext_vector_type(8)))  float            v8f;

// fp32 -> bf16, round to nearest even
__device__ inline u16 f2bf(float f) {
    unsigned u = __builtin_bit_cast(unsigned, f);
    unsigned r = u + 0x7FFFu + ((u >> 16) & 1u);
    return (u16)(r >> 16);
}

__device__ inline v8f wmma_bf16(v16bf a, v16bf b, v8f c) {
    // D = A(16x32 bf16) x B(32x16 bf16) + C(16x16 f32)
    return __builtin_amdgcn_wmma_f32_16x16x32_bf16(
        /*neg_a=*/false, a, /*neg_b=*/false, b,
        /*c_mod=*/(short)0, c, /*reuse_a=*/false, /*reuse_b=*/false);
}

// 16-bit A/B fragment (ISA 7.12.2): lane group kof = (lane&16)?8:0, and the
// lane's 16 values live at k = kof + {0..7, 16..23}  -> two aligned 16B loads.
__device__ inline v16bf frag_contig(const u16* p) {
    union { uint4 q[2]; v16bf v; } u;
    u.q[0] = *(const uint4*)(p);
    u.q[1] = *(const uint4*)(p + 16);
    return u.v;
}

__device__ inline v16bf frag_zero() {
    v16u z = {};
    return __builtin_bit_cast(v16bf, z);
}

// ---- CDNA5 async copy global->LDS (ASYNCcnt path) ----
__device__ inline void async_copy16(void* lds, const void* g) {
    unsigned loff = (unsigned)(unsigned long long)lds;  // low 32b = LDS offset
    asm volatile("global_load_async_to_lds_b128 %0, %1, off"
                 :: "v"(loff), "v"(g) : "memory");
}
__device__ inline void wait_async_le4() { asm volatile("s_wait_asynccnt 0x4" ::: "memory"); }
__device__ inline void wait_async_0()   { asm volatile("s_wait_asynccnt 0x0" ::: "memory"); }

// ---------------------------------------------------------------------------
// Tiled transpose + fp32->bf16: in [K][N] (per layer) -> out [N][K]
// grid: (N/32, K/32, LNUM), block 256 (32x8)
// ---------------------------------------------------------------------------
__global__ __launch_bounds__(256)
void convT_bf16(const float* __restrict__ in, u16* __restrict__ out, int K, int N) {
    __shared__ u16 tile[32][33];
    const float* src = in + (size_t)blockIdx.z * K * N;
    u16* dst = out + (size_t)blockIdx.z * K * N;
    const int k0 = blockIdx.y * 32, n0 = blockIdx.x * 32;
    const int tx = threadIdx.x & 31, ty = threadIdx.x >> 5;
    for (int r = ty; r < 32; r += 8)
        tile[r][tx] = f2bf(src[(size_t)(k0 + r) * N + n0 + tx]);
    __syncthreads();
    for (int r = ty; r < 32; r += 8)
        dst[(size_t)(n0 + r) * K + k0 + tx] = tile[tx][r];
}

// ---------------------------------------------------------------------------
// positions: pos = cumsum(mask)*mask + 1  (per batch row block-scan)
// ---------------------------------------------------------------------------
__global__ __launch_bounds__(256)
void pos_kernel(const int* __restrict__ ids, int* __restrict__ pos) {
    __shared__ int sc[256];
    const int b = blockIdx.x, t = threadIdx.x;
    int run = 0;
    for (int base = 0; base < SEQ; base += 256) {
        int m = (ids[b * SEQ + base + t] != 0) ? 1 : 0;
        sc[t] = m;
        __syncthreads();
        for (int off = 1; off < 256; off <<= 1) {
            int v = (t >= off) ? sc[t - off] : 0;
            __syncthreads();
            sc[t] += v;
            __syncthreads();
        }
        pos[b * SEQ + base + t] = (run + sc[t]) * m + 1;
        run += sc[255];
        __syncthreads();
    }
}

// ---------------------------------------------------------------------------
// embedding gather + LayerNorm -> h (f32) and h (bf16)
// ---------------------------------------------------------------------------
__global__ __launch_bounds__(256)
void embed_ln_kernel(const int* __restrict__ ids, const int* __restrict__ pos,
                     const float* __restrict__ wemb, const float* __restrict__ pemb,
                     const float* __restrict__ temb, const float* __restrict__ gs,
                     const float* __restrict__ gb, float* __restrict__ h,
                     u16* __restrict__ hbf) {
    __shared__ float red[256];
    const int row = blockIdx.x, t = threadIdx.x;
    const int id = ids[row], p = pos[row];
    float v[3];
#pragma unroll
    for (int k = 0; k < 3; ++k) {
        int d = t + k * 256;
        v[k] = wemb[(size_t)id * DDIM + d] + pemb[(size_t)p * DDIM + d] + temb[d];
    }
    red[t] = v[0] + v[1] + v[2];
    __syncthreads();
    for (int off = 128; off; off >>= 1) { if (t < off) red[t] += red[t + off]; __syncthreads(); }
    float mean = red[0] * (1.0f / DDIM);
    __syncthreads();
    float d0 = v[0] - mean, d1 = v[1] - mean, d2 = v[2] - mean;
    red[t] = d0 * d0 + d1 * d1 + d2 * d2;
    __syncthreads();
    for (int off = 128; off; off >>= 1) { if (t < off) red[t] += red[t + off]; __syncthreads(); }
    float inv = rsqrtf(red[0] * (1.0f / DDIM) + 1e-5f);
    float dv[3] = { d0, d1, d2 };
#pragma unroll
    for (int k = 0; k < 3; ++k) {
        int d = t + k * 256;
        float o = dv[k] * inv * gs[d] + gb[d];
        h[(size_t)row * DDIM + d] = o;
        hbf[(size_t)row * DDIM + d] = f2bf(o);
    }
}

// ---------------------------------------------------------------------------
// LayerNorm of x -> y (f32) and y (bf16)
// ---------------------------------------------------------------------------
__global__ __launch_bounds__(256)
void ln_kernel(const float* __restrict__ x, const float* __restrict__ gs,
               const float* __restrict__ gb, float* __restrict__ y,
               u16* __restrict__ ybf) {
    __shared__ float red[256];
    const int row = blockIdx.x, t = threadIdx.x;
    const float* xr = x + (size_t)row * DDIM;
    float v0 = xr[t], v1 = xr[t + 256], v2 = xr[t + 512];
    red[t] = v0 + v1 + v2;
    __syncthreads();
    for (int off = 128; off; off >>= 1) { if (t < off) red[t] += red[t + off]; __syncthreads(); }
    float mean = red[0] * (1.0f / DDIM);
    __syncthreads();
    float d0 = v0 - mean, d1 = v1 - mean, d2 = v2 - mean;
    red[t] = d0 * d0 + d1 * d1 + d2 * d2;
    __syncthreads();
    for (int off = 128; off; off >>= 1) { if (t < off) red[t] += red[t + off]; __syncthreads(); }
    float inv = rsqrtf(red[0] * (1.0f / DDIM) + 1e-5f);
    float dv[3] = { d0, d1, d2 };
#pragma unroll
    for (int k = 0; k < 3; ++k) {
        int d = t + k * 256;
        float o = dv[k] * inv * gs[d] + gb[d];
        y[(size_t)row * DDIM + d] = o;
        ybf[(size_t)row * DDIM + d] = f2bf(o);
    }
}

// ---------------------------------------------------------------------------
// WMMA GEMM: C[MxN] = A_bf16[MxK] @ W[KxN] (+bias); W supplied TRANSPOSED
// as WT[NxK] so both LDS tiles stage with plain aligned 16B copies.
// Double-buffered async global->LDS staging (ASYNCcnt), K-step 32.
// Block tile 128x128, 8 waves each owning a 32x64 sub-tile.
//   EPI 0: outF = res + acc + bias              (fp32, residual add)
//   EPI 1: outB = bf16((acc + bias) * scale)    (q/k projections)
//   EPI 2: outB = bf16(gelu(acc + bias))        (FFN up-proj)
//   EPI 3: outB[n][m] = bf16(acc + bias)        (v projection, feature-major)
// ---------------------------------------------------------------------------
template <int EPI>
__global__ __launch_bounds__(256)
void gemm_wmma(const u16* __restrict__ A, const u16* __restrict__ WT,
               const float* __restrict__ bias, const float* __restrict__ res,
               float* __restrict__ outF, u16* __restrict__ outB,
               int M, int N, int K, float scale) {
    __shared__ __align__(16) u16 As[2][128 * 32];   // [m][k]
    __shared__ __align__(16) u16 Bs[2][128 * 32];   // [n][k]
    const int tid = threadIdx.x;
    const int lane = tid & 31, wid = tid >> 5;
    const int mBlk = blockIdx.y * 128, nBlk = blockIdx.x * 128;
    const int wm = (wid & 3) * 32;
    const int wn = (wid >> 2) * 64;
    const int kof = (lane & 16) ? 8 : 0;
    const int KT = K >> 5;

    // fixed per-thread staging chunks (2 x 16B per tile per matrix)
    const int e0 = tid * 8, e1 = (tid + 256) * 8;
    const int r0 = e0 >> 5, c0 = e0 & 31;
    const int r1 = e1 >> 5, c1 = e1 & 31;

    auto issue = [&](int buf, int k0) {
        async_copy16(&As[buf][r0 * 32 + c0], A  + (size_t)(mBlk + r0) * K + k0 + c0);
        async_copy16(&As[buf][r1 * 32 + c1], A  + (size_t)(mBlk + r1) * K + k0 + c1);
        async_copy16(&Bs[buf][r0 * 32 + c0], WT + (size_t)(nBlk + r0) * K + k0 + c0);
        async_copy16(&Bs[buf][r1 * 32 + c1], WT + (size_t)(nBlk + r1) * K + k0 + c1);
    };

    v8f acc[2][4] = {};

    issue(0, 0);
    if (KT > 1) issue(1, 32);

    for (int t = 0; t < KT; ++t) {
        if (t + 1 < KT) wait_async_le4();   // oldest tile's 4 copies landed
        else            wait_async_0();
        __syncthreads();

        const u16* Ab = As[t & 1];
        const u16* Bb = Bs[t & 1];
        v16bf af[2], bfr[4];
#pragma unroll
        for (int i = 0; i < 2; ++i)
            af[i] = frag_contig(Ab + (wm + i * 16 + (lane & 15)) * 32 + kof);
#pragma unroll
        for (int j = 0; j < 4; ++j)
            bfr[j] = frag_contig(Bb + (wn + j * 16 + (lane & 15)) * 32 + kof);
#pragma unroll
        for (int i = 0; i < 2; ++i)
#pragma unroll
            for (int j = 0; j < 4; ++j)
                acc[i][j] = wmma_bf16(af[i], bfr[j], acc[i][j]);

        __syncthreads();                    // everyone done reading buf (t&1)
        if (t + 2 < KT) issue(t & 1, (t + 2) * 32);
    }

    // epilogue: C layout = lane n=(lane&15), VGPR e -> m = e + 8*(lane>>4)
#pragma unroll
    for (int i = 0; i < 2; ++i) {
#pragma unroll
        for (int j = 0; j < 4; ++j) {
#pragma unroll
            for (int e = 0; e < 8; ++e) {
                int gm = mBlk + wm + i * 16 + e + ((lane >> 4) << 3);
                int gn = nBlk + wn + j * 16 + (lane & 15);
                float v = acc[i][j][e] + bias[gn];
                if (EPI == 0) {
                    size_t o = (size_t)gm * N + gn;
                    outF[o] = res[o] + v;
                } else if (EPI == 1) {
                    outB[(size_t)gm * N + gn] = f2bf(v * scale);
                } else if (EPI == 2) {
                    float g = 0.5f * v * (1.0f + erff(v * 0.70710678f));
                    outB[(size_t)gm * N + gn] = f2bf(g);
                } else {
                    outB[(size_t)gn * M + gm] = f2bf(v * scale);  // transposed
                }
            }
        }
    }
}

// ---------------------------------------------------------------------------
// Sliding-window attention, one block = (batch, head, chunk, 32-row q slab).
// Full 768-wide banded score strip in LDS (144KB of the 320KB WGP LDS),
// masked softmax in place, then probs @ v via WMMA. V is feature-major so
// phase-3 fragments are contiguous; the 32-key tile validity is uniform
// (window edges are multiples of 256).
// ---------------------------------------------------------------------------
__global__ __launch_bounds__(256)
void attn_kernel(const u16* __restrict__ qg, const u16* __restrict__ kg,
                 const u16* __restrict__ vT, const int* __restrict__ ids,
                 u16* __restrict__ ag) {
    extern __shared__ char smem[];
    float* Ssc = (float*)smem;                                    // 32 x 768 f32
    u16*   Sp  = (u16*)(smem + (size_t)32 * 768 * sizeof(float)); // 32 x 768 bf16
    __shared__ float redA[256];
    __shared__ float redB[256];

    int idx = blockIdx.x;
    const int qs = idx & 7;  idx >>= 3;   // 32-row slab within chunk
    const int c  = idx & 15; idx >>= 4;   // chunk
    const int hh = idx % NHEAD;
    const int b  = idx / NHEAD;

    const int tid = threadIdx.x, lane = tid & 31, wid = tid >> 5;
    const int kof = (lane & 16) ? 8 : 0;
    const int tq0 = c * WWIN + qs * 32;   // first q token of this slab

    // ---- phase 1: scores = q @ k^T over 3W=768 key window ----
    // hoist the 4 q fragments (shared across all 6 N-tiles of this wave)
    v16bf aq[2][2];
#pragma unroll
    for (int ks = 0; ks < 2; ++ks) {
        aq[ks][0] = frag_contig(qg + ((size_t)(b * SEQ + tq0 + (lane & 15))) * DDIM
                                   + hh * HDIM + ks * 32 + kof);
        aq[ks][1] = frag_contig(qg + ((size_t)(b * SEQ + tq0 + 16 + (lane & 15))) * DDIM
                                   + hh * HDIM + ks * 32 + kof);
    }
    for (int nt = wid * 6; nt < wid * 6 + 6; ++nt) {
        v8f acc0 = {}, acc1 = {};
        const int ki = nt * 16 + (lane & 15);
        const int kpos = c * WWIN + ki - WWIN;
        const bool kvalid = (kpos >= 0) && (kpos < SEQ);
#pragma unroll
        for (int ks = 0; ks < 2; ++ks) {   // HD=64 = 2 K-steps of 32
            v16bf bb = kvalid
                ? frag_contig(kg + ((size_t)(b * SEQ + kpos)) * DDIM + hh * HDIM + ks * 32 + kof)
                : frag_zero();
            acc0 = wmma_bf16(aq[ks][0], bb, acc0);
            acc1 = wmma_bf16(aq[ks][1], bb, acc1);
        }
#pragma unroll
        for (int e = 0; e < 8; ++e) {
            int m = e + ((lane >> 4) << 3);
            int kcol = nt * 16 + (lane & 15);
            Ssc[m * 768 + kcol]        = acc0[e];
            Ssc[(m + 16) * 768 + kcol] = acc1[e];
        }
    }
    __syncthreads();

    // ---- phase 2: band + pad mask, softmax (8 threads per row) ----
    {
        const int r = tid >> 3, part = tid & 7;
        const int qi = qs * 32 + r;   // row within chunk [0,256)
        float mx = -3.0e38f;
        for (int k2 = part * 96; k2 < part * 96 + 96; ++k2) {
            float sv = Ssc[r * 768 + k2];
            int kpos = c * WWIN + k2 - WWIN;
            bool valid = (k2 >= qi) && (k2 <= qi + 2 * WWIN) &&
                         (kpos >= 0) && (kpos < SEQ) && (ids[b * SEQ + kpos] != 0);
            if (!valid) sv = -1.0e9f;
            Ssc[r * 768 + k2] = sv;
            mx = fmaxf(mx, sv);
        }
        redA[tid] = mx;
        __syncthreads();
        mx = redA[r * 8];
#pragma unroll
        for (int i = 1; i < 8; ++i) mx = fmaxf(mx, redA[r * 8 + i]);
        float sum = 0.f;
        for (int k2 = part * 96; k2 < part * 96 + 96; ++k2) {
            float e = __expf(Ssc[r * 768 + k2] - mx);
            Ssc[r * 768 + k2] = e;
            sum += e;
        }
        redB[tid] = sum;
        __syncthreads();
        sum = 0.f;
#pragma unroll
        for (int i = 0; i < 8; ++i) sum += redB[r * 8 + i];
        float inv = 1.0f / sum;
        for (int k2 = part * 96; k2 < part * 96 + 96; ++k2)
            Sp[r * 768 + k2] = f2bf(Ssc[r * 768 + k2] * inv);
    }
    __syncthreads();

    // ---- phase 3: out = probs @ v ; 8 waves = 2 m-tiles x 4 n-tiles ----
    {
        const int mt = wid & 1, nt = wid >> 1;
        const int vrow = hh * HDIM + nt * 16 + (lane & 15);          // feature
        const u16* vbase = vT + (size_t)vrow * (BATCH * SEQ) + b * SEQ;
        v8f acc = {};
        for (int kt = 0; kt < 24; ++kt) {   // K = 768 window keys
            v16bf pa = frag_contig(Sp + (mt * 16 + (lane & 15)) * 768 + kt * 32 + kof);
            const int kpos0 = c * WWIN - WWIN + kt * 32;   // uniform over tile
            v16bf vb = (kpos0 >= 0 && kpos0 < SEQ)
                     ? frag_contig(vbase + kpos0 + kof)
                     : frag_zero();
            acc = wmma_bf16(pa, vb, acc);
        }
#pragma unroll
        for (int e = 0; e < 8; ++e) {
            int m = mt * 16 + e + ((lane >> 4) << 3);
            int tok = tq0 + m;
            ag[((size_t)(b * SEQ + tok)) * DDIM + hh * HDIM + nt * 16 + (lane & 15)] =
                f2bf(acc[e]);
        }
    }
}

// ---------------------------------------------------------------------------
// mean over tokens 1..S-1 (two-stage) and cls-token copy
// ---------------------------------------------------------------------------
__global__ __launch_bounds__(256)
void mean_part_kernel(const float* __restrict__ h, float* __restrict__ part) {
    const int b = blockIdx.x, ch = blockIdx.y;
    int t0 = ch * 256; if (t0 == 0) t0 = 1;
    const int t1 = (ch + 1) * 256;
    for (int d = threadIdx.x; d < DDIM; d += 256) {
        float s = 0.f;
        for (int t = t0; t < t1; ++t)
            s += h[((size_t)b * SEQ + t) * DDIM + d];
        part[((size_t)b * 16 + ch) * DDIM + d] = s;
    }
}

__global__ __launch_bounds__(256)
void mean_final_kernel(const float* __restrict__ part, const float* __restrict__ h,
                       float* __restrict__ mean, float* __restrict__ outCls) {
    const int b = blockIdx.x;
    for (int d = threadIdx.x; d < DDIM; d += 256) {
        float s = 0.f;
#pragma unroll
        for (int ch = 0; ch < 16; ++ch)
            s += part[((size_t)b * 16 + ch) * DDIM + d];
        mean[b * DDIM + d] = s * (1.0f / (SEQ - 1));
        outCls[b * DDIM + d] = h[(size_t)b * SEQ * DDIM + d];
    }
}

// result[b][j] = mean[b] . cls_W[j] + cls_b[j]
__global__ __launch_bounds__(256)
void cls_kernel(const float* __restrict__ mean, const float* __restrict__ cw,
                const float* __restrict__ cb, float* __restrict__ out) {
    __shared__ float red[256];
    const int o = blockIdx.x, b = o >> 1, j = o & 1, t = threadIdx.x;
    float p = 0.f;
    for (int d = t; d < DDIM; d += 256) p += mean[b * DDIM + d] * cw[j * DDIM + d];
    red[t] = p;
    __syncthreads();
    for (int off = 128; off; off >>= 1) { if (t < off) red[t] += red[t + off]; __syncthreads(); }
    if (t == 0) out[b * 2 + j] = red[0] + cb[j];
}

// ---------------------------------------------------------------------------
extern "C" void kernel_launch(void* const* d_in, const int* in_sizes, int n_in,
                              void* d_out, int out_size, void* d_ws, size_t ws_size,
                              hipStream_t stream) {
    (void)in_sizes; (void)n_in; (void)out_size; (void)ws_size;

    const int*   ids  = (const int*)  d_in[0];
    const float* wemb = (const float*)d_in[1];
    const float* pemb = (const float*)d_in[2];
    const float* temb = (const float*)d_in[3];
    const float* lnes = (const float*)d_in[4];
    const float* lneb = (const float*)d_in[5];
    const float* Wq   = (const float*)d_in[6];
    const float* bq   = (const float*)d_in[7];
    const float* Wk   = (const float*)d_in[8];
    const float* bk   = (const float*)d_in[9];
    const float* Wv   = (const float*)d_in[10];
    const float* bv   = (const float*)d_in[11];
    const float* Wo   = (const float*)d_in[12];
    const float* bo   = (const float*)d_in[13];
    const float* ln1s = (const float*)d_in[14];
    const float* ln1b = (const float*)d_in[15];
    const float* W1   = (const float*)d_in[16];
    const float* b1   = (const float*)d_in[17];
    const float* W2   = (const float*)d_in[18];
    const float* b2   = (const float*)d_in[19];
    const float* ln2s = (const float*)d_in[20];
    const float* ln2b = (const float*)d_in[21];
    const float* clsW = (const float*)d_in[22];
    const float* clsb = (const float*)d_in[23];
    float* out = (float*)d_out;

    const size_t DD = (size_t)DDIM * DDIM;
    const size_t DF = (size_t)DDIM * FFDIM;
    const size_t MT = (size_t)BATCH * SEQ;   // 8192 token rows

    char* p = (char*)d_ws;
    auto carve = [&](size_t bytes) -> void* {
        void* r = (void*)p;
        p += (bytes + 255) & ~(size_t)255;
        return r;
    };
    u16*   WqT  = (u16*)carve(LNUM * DD * 2);   // all weights pre-transposed [N][K]
    u16*   WkT  = (u16*)carve(LNUM * DD * 2);
    u16*   WvT  = (u16*)carve(LNUM * DD * 2);
    u16*   WoT  = (u16*)carve(LNUM * DD * 2);
    u16*   W1T  = (u16*)carve(LNUM * DF * 2);
    u16*   W2T  = (u16*)carve(LNUM * DF * 2);
    float* hF   = (float*)carve(MT * DDIM * 4);
    float* tmpF = (float*)carve(MT * DDIM * 4);
    u16*   hB   = (u16*)carve(MT * DDIM * 2);
    u16*   qB   = (u16*)carve(MT * DDIM * 2);
    u16*   kB   = (u16*)carve(MT * DDIM * 2);
    u16*   vTb  = (u16*)carve(MT * DDIM * 2);   // v, feature-major [768][8192]
    u16*   aB   = (u16*)carve(MT * DDIM * 2);
    u16*   fB   = (u16*)carve(MT * FFDIM * 2);
    int*   posI = (int*)carve(MT * 4);
    float* mprt = (float*)carve((size_t)BATCH * 16 * DDIM * 4);
    float* mF   = (float*)carve((size_t)BATCH * DDIM * 4);

    // one-time (per launch) weight transpose + bf16 downconvert
    convT_bf16<<<dim3(DDIM / 32, DDIM / 32, LNUM), 256, 0, stream>>>(Wq, WqT, DDIM, DDIM);
    convT_bf16<<<dim3(DDIM / 32, DDIM / 32, LNUM), 256, 0, stream>>>(Wk, WkT, DDIM, DDIM);
    convT_bf16<<<dim3(DDIM / 32, DDIM / 32, LNUM), 256, 0, stream>>>(Wv, WvT, DDIM, DDIM);
    convT_bf16<<<dim3(DDIM / 32, DDIM / 32, LNUM), 256, 0, stream>>>(Wo, WoT, DDIM, DDIM);
    convT_bf16<<<dim3(FFDIM / 32, DDIM / 32, LNUM), 256, 0, stream>>>(W1, W1T, DDIM, FFDIM);
    convT_bf16<<<dim3(DDIM / 32, FFDIM / 32, LNUM), 256, 0, stream>>>(W2, W2T, FFDIM, DDIM);

    pos_kernel<<<BATCH, 256, 0, stream>>>(ids, posI);
    embed_ln_kernel<<<(int)MT, 256, 0, stream>>>(ids, posI, wemb, pemb, temb,
                                                 lnes, lneb, hF, hB);

    dim3 gD(DDIM / 128, (int)(MT / 128));    // (6, 64)
    dim3 gF(FFDIM / 128, (int)(MT / 128));   // (24, 64)
    const int attn_blocks = BATCH * NHEAD * (SEQ / WWIN) * (WWIN / 32);  // 3072
    const size_t attn_smem = (size_t)32 * 768 * 4 + (size_t)32 * 768 * 2; // 144 KB

    for (int l = 0; l < LNUM; ++l) {
        // q/k/v projections (q pre-scaled by 1/sqrt(HD); v stored feature-major)
        gemm_wmma<1><<<gD, 256, 0, stream>>>(hB, WqT + l * DD, bq + l * DDIM,
                                             nullptr, nullptr, qB,
                                             (int)MT, DDIM, DDIM, 0.125f);
        gemm_wmma<1><<<gD, 256, 0, stream>>>(hB, WkT + l * DD, bk + l * DDIM,
                                             nullptr, nullptr, kB,
                                             (int)MT, DDIM, DDIM, 1.0f);
        gemm_wmma<3><<<gD, 256, 0, stream>>>(hB, WvT + l * DD, bv + l * DDIM,
                                             nullptr, nullptr, vTb,
                                             (int)MT, DDIM, DDIM, 1.0f);
        // banded attention
        attn_kernel<<<attn_blocks, 256, attn_smem, stream>>>(qB, kB, vTb, ids, aB);
        // output projection + residual, then LN1
        gemm_wmma<0><<<gD, 256, 0, stream>>>(aB, WoT + l * DD, bo + l * DDIM,
                                             hF, tmpF, nullptr,
                                             (int)MT, DDIM, DDIM, 1.0f);
        ln_kernel<<<(int)MT, 256, 0, stream>>>(tmpF, ln1s + l * DDIM, ln1b + l * DDIM, hF, hB);
        // FFN
        gemm_wmma<2><<<gF, 256, 0, stream>>>(hB, W1T + l * DF, b1 + l * FFDIM,
                                             nullptr, nullptr, fB,
                                             (int)MT, FFDIM, DDIM, 1.0f);
        gemm_wmma<0><<<gD, 256, 0, stream>>>(fB, W2T + l * DF, b2 + l * DDIM,
                                             hF, tmpF, nullptr,
                                             (int)MT, DDIM, FFDIM, 1.0f);
        ln_kernel<<<(int)MT, 256, 0, stream>>>(tmpF, ln2s + l * DDIM, ln2b + l * DDIM, hF, hB);
    }

    // outputs: result (2x2) at out[0..3], cls_token (2x768) at out[4..1539]
    mean_part_kernel<<<dim3(BATCH, 16), 256, 0, stream>>>(hF, mprt);
    mean_final_kernel<<<BATCH, 256, 0, stream>>>(mprt, hF, mF, out + 4);
    cls_kernel<<<BATCH * 2, 256, 0, stream>>>(mF, clsW, clsb, out);
}
